// LSTM_23459111371000
// MI455X (gfx1250) — matrix operand order repaired and checked by reading, not compiled, not run on hardware
//
#include <hip/hip_runtime.h>
#include <stdint.h>

#define B_  1024
#define S_  20
#define H_  2048
#define L_  2000
#define PL_ 30
#define G4_ (4*H_)

typedef __bf16 bfx16 __attribute__((ext_vector_type(16)));
typedef __bf16 bfx8  __attribute__((ext_vector_type(8)));
typedef float  fx8   __attribute__((ext_vector_type(8)));

__device__ __forceinline__ unsigned short f2bf(float f){
  union { float f; unsigned int u; } x; x.f = f;
  unsigned int r = x.u + 0x7FFFu + ((x.u >> 16) & 1u);
  return (unsigned short)(r >> 16);
}
__device__ __forceinline__ float sigmf(float x){ return 1.0f / (1.0f + __expf(-x)); }

// ---------------------------------------------------------------- prep / init
__global__ void prep_kernel(const float* __restrict__ in,
                            float* __restrict__ fc, float* __restrict__ xs)
{
  int idx = blockIdx.x * blockDim.x + threadIdx.x;
  if (idx >= B_ * S_) return;
  int b = idx / S_, t = idx - b * S_;
  float f0 = in[(size_t)b * S_ * 2 + 0], f1 = in[(size_t)b * S_ * 2 + 1];
  float v0 = in[((size_t)b * S_ + t) * 2 + 0], v1 = in[((size_t)b * S_ + t) * 2 + 1];
  xs[((size_t)t * B_ + b) * 2 + 0] = v0 - f0;
  xs[((size_t)t * B_ + b) * 2 + 1] = v1 - f1;
  if (t == 0) { fc[b * 2] = f0; fc[b * 2 + 1] = f1; }
}

__global__ void zero_kernel(float* __restrict__ c, unsigned short* __restrict__ h0)
{
  int i = blockIdx.x * blockDim.x + threadIdx.x;
  if (i < B_ * H_) { c[i] = 0.0f; h0[i] = 0; }
}

__global__ void conv_kernel(const float* __restrict__ w, unsigned short* __restrict__ wb)
{
  int i = (blockIdx.x * blockDim.x + threadIdx.x) * 4;
  if (i + 3 < G4_ * H_) {
    wb[i + 0] = f2bf(w[i + 0]);
    wb[i + 1] = f2bf(w[i + 1]);
    wb[i + 2] = f2bf(w[i + 2]);
    wb[i + 3] = f2bf(w[i + 3]);
  }
}

// ------------------------------------------------- nearest-lane augmentation
__global__ void augment_kernel(const float* __restrict__ pos,
                               const float* __restrict__ lanes,
                               const float* __restrict__ lnorm,
                               const float* __restrict__ fc,
                               float* __restrict__ xb)
{
  int wid  = (blockIdx.x * blockDim.x + threadIdx.x) >> 5;
  int lane = threadIdx.x & 31;
  if (wid >= B_) return;
  int s = wid >> 6;                       // scene = b / 64 agents
  float px = pos[wid * 2 + 0], py = pos[wid * 2 + 1];
  const float* lp = lanes + (size_t)s * L_ * 2;
  float best = 3.4e38f; int bi = 0;
  for (int i = lane; i < L_; i += 32) {
    float dx = px - lp[i * 2 + 0];
    float dy = py - lp[i * 2 + 1];
    float d  = dx * dx + dy * dy;
    if (d < best) { best = d; bi = i; }
  }
#pragma unroll
  for (int off = 16; off; off >>= 1) {
    float ob = __shfl_down(best, off, 32);
    int   oi = __shfl_down(bi,   off, 32);
    if (ob < best || (ob == best && oi < bi)) { best = ob; bi = oi; }
  }
  if (lane == 0) {
    const float* np = lnorm + (size_t)s * L_ * 2;
    float nlx = lp[bi * 2 + 0], nly = lp[bi * 2 + 1];
    float nnx = np[bi * 2 + 0], nny = np[bi * 2 + 1];
    float fx = fc[wid * 2 + 0], fy = fc[wid * 2 + 1];
    float* o = xb + wid * 6;
    o[0] = px;                      o[1] = py;
    o[2] = nlx - 0.5f * nnx - fx;   o[3] = nly - 0.5f * nny - fy;
    o[4] = nlx + 0.5f * nnx - fx;   o[5] = nly + 0.5f * nny - fy;
  }
}

// ---------------------------------------------- fused WMMA GEMM + LSTM cell
// block: 8 waves, owns 64 batch rows x 64 hidden cols, all 4 gate quadrants.
// wave w -> (wm2 = w>>2 in {0,1}: 32-row half; wn = w&3: 16-col subtile).
// Each wave: 2 M-tiles x 4 quadrants = 8 accumulators; B fragments are
// reused across both M-tiles (24 B loaded per lane per WMMA vs 40 before).
__global__ void __launch_bounds__(256)
cell_kernel(const unsigned short* __restrict__ hin,   // B x H bf16 (prev h)
            const unsigned short* __restrict__ wbf,   // 4H x H bf16 (W_hh)
            const float* __restrict__ wih,            // 4H x 6
            const float* __restrict__ bih,
            const float* __restrict__ bhh,
            const float* __restrict__ xb,             // B x 6 (augmented x)
            float* __restrict__ cst,                  // B x H cell state
            float* __restrict__ hf,                   // B x H new h (fp32)
            unsigned short* __restrict__ hout)        // B x H new h (bf16)
{
  __shared__ float xsh[64][6];
  __shared__ float bsh[4][64];

  const int tid  = threadIdx.x;
  const int lane = tid & 31;
  const int w    = tid >> 5;
  const int wm2  = w >> 2;              // 0..1 : which 32-row half
  const int wn   = w & 3;               // 0..3 : 16-col subtile
  const int rb   = blockIdx.x >> 5;     // B_/64 = 16 row blocks
  const int cb   = blockIdx.x & 31;     // H_/64 = 32 col blocks
  const int m0   = rb * 64;
  const int j0   = cb * 64;
  const int ln   = lane & 15;
  const int hh   = lane >> 4;

  for (int i = tid; i < 64 * 6; i += 256) {
    int r = i / 6, k = i - r * 6;
    xsh[r][k] = xb[(m0 + r) * 6 + k];
  }
  { int q = tid >> 6, n = tid & 63;
    bsh[q][n] = bih[q * H_ + j0 + n] + bhh[q * H_ + j0 + n]; }
  __syncthreads();

  // A fragments: two 16-row tiles of prev-h (contiguous k order, same as B:
  // lane-half 0 holds K {0..15}, half 1 holds K {16..31})
  const unsigned short* aRow0 = hin + (size_t)(m0 + wm2 * 32 + ln) * H_;
  const unsigned short* aRow1 = aRow0 + (size_t)16 * H_;
  // B fragments: W_hh row = gate column; lane-half selects K-half
  const unsigned short* bR[4];
#pragma unroll
  for (int q = 0; q < 4; q++)
    bR[q] = wbf + (size_t)(q * H_ + j0 + wn * 16 + ln) * H_ + hh * 16;

  fx8 z = {0.f, 0.f, 0.f, 0.f, 0.f, 0.f, 0.f, 0.f};
  fx8 acc[2][4] = { { z, z, z, z }, { z, z, z, z } };

  for (int k0 = 0; k0 < H_; k0 += 32) {
    bfx8 a0lo = *(const bfx8*)(aRow0 + k0 + hh * 16);
    bfx8 a0hi = *(const bfx8*)(aRow0 + k0 + 8 + hh * 16);
    bfx8 a1lo = *(const bfx8*)(aRow1 + k0 + hh * 16);
    bfx8 a1hi = *(const bfx8*)(aRow1 + k0 + 8 + hh * 16);
    bfx16 a0, a1;
#pragma unroll
    for (int i = 0; i < 8; i++) {
      a0[i] = a0lo[i]; a0[8 + i] = a0hi[i];
      a1[i] = a1lo[i]; a1[8 + i] = a1hi[i];
    }
#pragma unroll
    for (int q = 0; q < 4; q++) {
      bfx16 b = *(const bfx16*)(bR[q] + k0);
      acc[0][q] = __builtin_amdgcn_wmma_f32_16x16x32_bf16(
          false, a0, false, b, (short)0, acc[0][q], false, false);
      acc[1][q] = __builtin_amdgcn_wmma_f32_16x16x32_bf16(
          false, a1, false, b, (short)0, acc[1][q], false, false);
    }
  }

  // x @ W_ih^T epilogue: column (hence W_ih row) is fixed per lane
  float wv[4][6];
#pragma unroll
  for (int q = 0; q < 4; q++) {
    const float* wp = wih + (size_t)(q * H_ + j0 + wn * 16 + ln) * 6;
#pragma unroll
    for (int k = 0; k < 6; k++) wv[q][k] = wp[k];
  }

  const int nl = wn * 16 + ln;
#pragma unroll
  for (int mt = 0; mt < 2; mt++) {
#pragma unroll
    for (int v = 0; v < 8; v++) {
      int ml = wm2 * 32 + mt * 16 + v + 8 * hh;  // C/D layout: half -> M+8
      int m  = m0 + ml;
      int j  = j0 + nl;
      float g0 = acc[mt][0][v] + bsh[0][nl];
      float g1 = acc[mt][1][v] + bsh[1][nl];
      float g2 = acc[mt][2][v] + bsh[2][nl];
      float g3 = acc[mt][3][v] + bsh[3][nl];
#pragma unroll
      for (int k = 0; k < 6; k++) {
        float xv = xsh[ml][k];
        g0 += xv * wv[0][k]; g1 += xv * wv[1][k];
        g2 += xv * wv[2][k]; g3 += xv * wv[3][k];
      }
      size_t off = (size_t)m * H_ + j;
      float cn = sigmf(g1) * cst[off] + sigmf(g0) * tanhf(g2);
      float hn = sigmf(g3) * tanhf(cn);
      cst[off] = cn;
      hf[off]  = hn;
      hout[off] = f2bf(hn);
    }
  }
}

// ------------------------------------------------ decoder output projection
__global__ void outproj_kernel(const float* __restrict__ hf,
                               const float* __restrict__ wout,  // 2 x H
                               const float* __restrict__ bout,
                               const float* __restrict__ fc,
                               float* __restrict__ pos,
                               float* __restrict__ dout, int t)
{
  int wid  = (blockIdx.x * blockDim.x + threadIdx.x) >> 5;
  int lane = threadIdx.x & 31;
  if (wid >= B_) return;
  const float* hp = hf + (size_t)wid * H_;
  float s0 = 0.f, s1 = 0.f;
  for (int k = lane; k < H_; k += 32) {
    float hv = hp[k];
    s0 += hv * wout[k];
    s1 += hv * wout[H_ + k];
  }
#pragma unroll
  for (int off = 16; off; off >>= 1) {
    s0 += __shfl_down(s0, off, 32);
    s1 += __shfl_down(s1, off, 32);
  }
  if (lane == 0) {
    float o0 = s0 + bout[0], o1 = s1 + bout[1];
    pos[wid * 2 + 0] = o0;
    pos[wid * 2 + 1] = o1;
    size_t oo = ((size_t)wid * PL_ + t) * 2;
    dout[oo + 0] = o0 + fc[wid * 2 + 0];
    dout[oo + 1] = o1 + fc[wid * 2 + 1];
  }
}

// ---------------------------------------------------------------------------
extern "C" void kernel_launch(void* const* d_in, const int* in_sizes, int n_in,
                              void* d_out, int out_size, void* d_ws, size_t ws_size,
                              hipStream_t stream)
{
  const float* in_seq = (const float*)d_in[0];
  const float* lanes  = (const float*)d_in[1];
  const float* lnorm  = (const float*)d_in[2];
  const float* wih    = (const float*)d_in[3];
  const float* whh    = (const float*)d_in[4];
  const float* bih    = (const float*)d_in[5];
  const float* bhh    = (const float*)d_in[6];
  const float* wout   = (const float*)d_in[7];
  const float* bout   = (const float*)d_in[8];
  float* dout = (float*)d_out;

  char* ws = (char*)d_ws;
  size_t off = 0;
  auto alloc = [&](size_t bytes) -> void* {
    void* p = ws + off;
    off += (bytes + 255) & ~(size_t)255;
    return p;
  };
  unsigned short* wbf  = (unsigned short*)alloc((size_t)G4_ * H_ * 2); // 32 MB
  unsigned short* hbf0 = (unsigned short*)alloc((size_t)B_ * H_ * 2);  //  4 MB
  unsigned short* hbf1 = (unsigned short*)alloc((size_t)B_ * H_ * 2);  //  4 MB
  float* cst = (float*)alloc((size_t)B_ * H_ * 4);                     //  8 MB
  float* hf  = (float*)alloc((size_t)B_ * H_ * 4);                     //  8 MB
  float* fc  = (float*)alloc((size_t)B_ * 2 * 4);
  float* xs  = (float*)alloc((size_t)S_ * B_ * 2 * 4);
  float* xb  = (float*)alloc((size_t)B_ * 6 * 4);
  float* pos = (float*)alloc((size_t)B_ * 2 * 4);

  prep_kernel<<<(B_ * S_ + 255) / 256, 256, 0, stream>>>(in_seq, fc, xs);
  zero_kernel<<<(B_ * H_ + 255) / 256, 256, 0, stream>>>(cst, hbf0);
  conv_kernel<<<(G4_ * H_ / 4 + 255) / 256, 256, 0, stream>>>(whh, wbf);

  unsigned short* hb[2] = { hbf0, hbf1 };
  int p = 0;
  const int cellGrid = (B_ / 64) * (H_ / 64);   // 512 blocks
  for (int t = 0; t < S_; t++) {
    augment_kernel<<<B_ / 8, 256, 0, stream>>>(xs + (size_t)t * B_ * 2,
                                               lanes, lnorm, fc, xb);
    cell_kernel<<<cellGrid, 256, 0, stream>>>(
        hb[p], wbf, wih, bih, bhh, xb, cst, hf, hb[p ^ 1]);
    p ^= 1;
  }
  for (int t = 0; t < PL_; t++) {
    outproj_kernel<<<B_ / 8, 256, 0, stream>>>(hf, wout, bout, fc, pos, dout, t);
    if (t < PL_ - 1) {   // final cell's carry is never read by the reference
      augment_kernel<<<B_ / 8, 256, 0, stream>>>(pos, lanes, lnorm, fc, xb);
      cell_kernel<<<cellGrid, 256, 0, stream>>>(
          hb[p], wbf, wih, bih, bhh, xb, cst, hf, hb[p ^ 1]);
      p ^= 1;
    }
  }
}
